// Decoder_56075093017099
// MI455X (gfx1250) — compile-verified, matrix operand order checked by reading, not run on hardware
//
#include <hip/hip_runtime.h>
#include <hip/hip_bf16.h>
#include <stdint.h>

// ---------------------------------------------------------------------------
// 2-layer bidirectional LSTM (B=16, T=512, H=E=1024) for gfx1250.
// bf16 WMMA (16x16x32, f32 accum); batch(16) == one M tile.
// Weights repacked once per call into per-lane B-fragment layout (84MB bf16,
// L2-resident). Layers software-pipelined: ONE kernel per step (513 launches),
// blocks 0..127 = layer0 step t, blocks 128..255 = layer1 step t-1.
// States kept in f32; only matmul operands rounded to bf16.
// ---------------------------------------------------------------------------

typedef unsigned short u16;
typedef __attribute__((ext_vector_type(16))) __bf16 v16bf;
typedef __attribute__((ext_vector_type(8)))  float  v8f;

union Frag { v16bf v; uint4 q[2]; };

#define Hd   1024
#define Bd   16
#define Td   512
#define BOS  1

__device__ __forceinline__ u16 f2bf(float x) {
    uint32_t u = __float_as_uint(x);
    u += 0x7fffu + ((u >> 16) & 1u);   // round to nearest even
    return (u16)(u >> 16);
}

__device__ __forceinline__ uint2 f4_to_bf4(float4 v) {
    uint2 r;
    r.x = (uint32_t)f2bf(v.x) | ((uint32_t)f2bf(v.y) << 16);
    r.y = (uint32_t)f2bf(v.z) | ((uint32_t)f2bf(v.w) << 16);
    return r;
}

__device__ __forceinline__ float sigf(float x) {
    return 1.0f / (1.0f + __expf(-x));
}

// --- weight repack: f32 [4096, Kpart] -> bf16 fragment layout -------------
// dst layout (per cell): [ntile(256)][kt(KT)][lane(32)][16 bf16]
// 16-bit A/B layout per ISA: lane (hi*16+row); VGPR v, half h -> K index:
//   K = (v%4)*2 + (v/4)*16 + hi*8 + h
__global__ void pack_w_kernel(const float* __restrict__ W, u16* __restrict__ dst,
                              int Kpart, int KT, int kt0) {
    int idx = blockIdx.x * blockDim.x + threadIdx.x;
    int total = 4096 * Kpart;
    if (idx >= total) return;
    int n = idx / Kpart, k = idx % Kpart;
    int ntile = n >> 4, row = n & 15;
    int ktp = k >> 5, kk = k & 31;
    int hi = (kk >> 3) & 1;
    int tt = kk - (hi ? 8 : 0);                 // in {0..7} U {16..23}
    int v  = (tt < 8) ? (tt >> 1) : (4 + ((tt - 16) >> 1));
    int h  = tt & 1;
    int lane = hi * 16 + row;
    size_t o = (((size_t)ntile * KT + kt0 + ktp) * 32 + lane) * 16 + (2 * v + h);
    dst[o] = f2bf(W[(size_t)n * Kpart + k]);
}

__global__ void pack_bias_kernel(const float* __restrict__ bih,
                                 const float* __restrict__ bhh,
                                 float* __restrict__ dst, int n) {
    int i = blockIdx.x * blockDim.x + threadIdx.x;
    if (i < n) dst[i] = bih[i] + bhh[i];
}

__global__ void zero_kernel(float* __restrict__ p, int n) {
    int i = blockIdx.x * blockDim.x + threadIdx.x;
    if (i < n) p[i] = 0.0f;
}

// --- cell body: one layer, both directions, one time step ------------------
// bx in [0,128): d = bx>>6 direction, jb = bx&63 j-block of 16 hidden units.
// 128 threads = 4 waves; wave g owns gate g (i,f,g,o).
template <int KTOT, int LAYER>
__device__ __forceinline__
void cell_body(int bx, u16* smem,
               const int*   __restrict__ inputs,   // [16,512]
               const float* __restrict__ emb,      // [V,1024]
               const u16*   __restrict__ Wpack,    // [2][256][KT][32][16]
               const float* __restrict__ bias,     // [2][4096]
               const float* __restrict__ Hread,    // l0: prev H0 [16,2048]; l1: x1 [16,2048]
               const float* __restrict__ Hrec,     // l1 only: prev H1 [2][16][1024]
               float*       __restrict__ Hwrite,   // l0: H0' [16,2048]; l1: H1' [2][16][1024]
               float*       __restrict__ Cst,      // [2][16][1024]
               float*       __restrict__ out,      // l1: [16,512,2048]
               int t) {
    constexpr int KT = KTOT / 32;
    const int d  = bx >> 6;
    const int jb = bx & 63;
    u16* As = smem;                               // [16][KTOT] bf16

    // ---- stage A = [x | h] into LDS as bf16 (float4 -> 4x bf16) ----
    for (int i = threadIdx.x; i < Bd * (KTOT / 4); i += blockDim.x) {
        int row = i / (KTOT / 4);
        int k4  = (i % (KTOT / 4)) * 4;
        float4 v;
        if (LAYER == 0) {
            if (k4 < 1024) {
                int tok = (t == 0) ? BOS : inputs[row * Td + t];
                v = *(const float4*)(emb + (size_t)tok * 1024 + k4);
            } else {
                v = *(const float4*)(Hread + row * 2048 + d * 1024 + (k4 - 1024));
            }
        } else {
            if (k4 < 2048) {
                v = *(const float4*)(Hread + row * 2048 + k4);
            } else {
                v = *(const float4*)(Hrec + ((size_t)d * Bd + row) * 1024 + (k4 - 2048));
            }
        }
        *(uint2*)(As + row * KTOT + k4) = f4_to_bf4(v);
    }
    __syncthreads();

    // ---- WMMA: gate g, 16 output columns, full K sweep ----
    const int lane   = threadIdx.x & 31;
    const int g      = threadIdx.x >> 5;          // gate index 0..3 (i,f,g,o)
    const int laneHi = lane >> 4;
    const int r16    = lane & 15;
    const int ntile  = g * 64 + jb;               // N tile within [0,4096)

    const u16* Wc = Wpack + ((size_t)d * 256 * KT + (size_t)ntile * KT) * 512;
    const float bv = bias[d * 4096 + g * 1024 + jb * 16 + r16];

    v8f acc;
#pragma unroll
    for (int i = 0; i < 8; ++i) acc[i] = bv;

    const uint4* arow = (const uint4*)(As + r16 * KTOT);  // 16B granules of row r16
    for (int kt = 0; kt < KT; ++kt) {
        Frag a, b;
        // A: lane's two contiguous 16B K-chunks (per 16-bit A layout)
        a.q[0] = arow[kt * 4 + laneHi];
        a.q[1] = arow[kt * 4 + 2 + laneHi];
        // B: prepacked, 32B contiguous per lane
        const uint4* bp = (const uint4*)(Wc + (size_t)kt * 512 + lane * 16);
        b.q[0] = bp[0];
        b.q[1] = bp[1];
        acc = __builtin_amdgcn_wmma_f32_16x16x32_bf16(
                  false, a.v, false, b.v, (short)0, acc, false, false);
    }

    // ---- exchange gate tiles through LDS ----
    float* Gs = (float*)(smem + Bd * KTOT);       // [4][16][16]
#pragma unroll
    for (int r = 0; r < 8; ++r)
        Gs[g * 256 + (r + laneHi * 8) * 16 + r16] = acc[r];
    __syncthreads();

    // ---- elementwise LSTM update (2 elements / thread) ----
    for (int p = threadIdx.x; p < 256; p += blockDim.x) {
        int m  = p >> 4;                          // batch row
        int jj = p & 15;
        float iG = Gs[0 * 256 + p];
        float fG = Gs[1 * 256 + p];
        float gG = Gs[2 * 256 + p];
        float oG = Gs[3 * 256 + p];
        int jglob = jb * 16 + jj;
        float* cp = Cst + ((size_t)d * Bd + m) * 1024 + jglob;
        float c2 = sigf(fG) * cp[0] + sigf(iG) * tanhf(gG);
        float h2 = sigf(oG) * tanhf(c2);
        cp[0] = c2;
        if (LAYER == 0) {
            Hwrite[m * 2048 + d * 1024 + jglob] = h2;   // x1 layout [16,2H]
        } else {
            Hwrite[((size_t)d * Bd + m) * 1024 + jglob] = h2;
            out[(((size_t)m * Td) + t) * 2048 + d * 1024 + jglob] = h2;
        }
    }
}

// --- pipelined step: layer0 @ t (blocks 0..127), layer1 @ t-1 (128..255) ---
__global__ __launch_bounds__(128)
void lstm_pipe_kernel(const int*   __restrict__ inputs,
                      const float* __restrict__ emb,
                      const u16*   __restrict__ Wl0,
                      const u16*   __restrict__ Wl1,
                      const float* __restrict__ bias0,
                      const float* __restrict__ bias1,
                      const float* __restrict__ H0r,  // layer0 reads (prev H0)
                      float*       __restrict__ H0w,  // layer0 writes
                      const float* __restrict__ X1,   // layer1 reads (H0 @ t-1)
                      const float* __restrict__ H1r,  // layer1 reads (prev H1)
                      float*       __restrict__ H1w,  // layer1 writes
                      float*       __restrict__ C0,
                      float*       __restrict__ C1,
                      float*       __restrict__ out,
                      int t) {
    extern __shared__ u16 smem[];
    if (blockIdx.x < 128) {
        if (t >= Td) return;                      // drain launch: no layer0 work
        cell_body<2048, 0>(blockIdx.x, smem, inputs, emb, Wl0, bias0,
                           H0r, (const float*)nullptr, H0w, C0, (float*)nullptr, t);
    } else {
        if (t == 0) return;                       // fill launch: no layer1 work yet
        cell_body<3072, 1>(blockIdx.x - 128, smem, inputs, emb, Wl1, bias1,
                           X1, H1r, H1w, C1, out, t - 1);
    }
}

// ---------------------------------------------------------------------------
extern "C" void kernel_launch(void* const* d_in, const int* in_sizes, int n_in,
                              void* d_out, int out_size, void* d_ws, size_t ws_size,
                              hipStream_t stream) {
    const int*   inputs  = (const int*)  d_in[0];
    const float* emb     = (const float*)d_in[1];
    const float* w_ih_l0 = (const float*)d_in[2];   // [2,4096,1024]
    const float* w_hh_l0 = (const float*)d_in[3];   // [2,4096,1024]
    const float* b_ih_l0 = (const float*)d_in[4];   // [2,4096]
    const float* b_hh_l0 = (const float*)d_in[5];
    const float* w_ih_l1 = (const float*)d_in[6];   // [2,4096,2048]
    const float* w_hh_l1 = (const float*)d_in[7];   // [2,4096,1024]
    const float* b_ih_l1 = (const float*)d_in[8];
    const float* b_hh_l1 = (const float*)d_in[9];
    float* out = (float*)d_out;

    // ---- workspace layout ----
    uint8_t* wsb = (uint8_t*)d_ws;
    size_t off = 0;
    auto take = [&](size_t bytes) -> uint8_t* {
        off = (off + 255) & ~(size_t)255;
        uint8_t* p = wsb + off;
        off += bytes;
        return p;
    };
    const size_t W0_ELEMS = (size_t)2 * 256 * 64 * 512;   // bf16 elems
    const size_t W1_ELEMS = (size_t)2 * 256 * 96 * 512;
    u16*   Wl0   = (u16*)  take(W0_ELEMS * 2);
    u16*   Wl1   = (u16*)  take(W1_ELEMS * 2);
    float* bias0 = (float*)take(2 * 4096 * 4);
    float* bias1 = (float*)take(2 * 4096 * 4);
    // zeroed region: H0[0], H1[0], C0, C1 contiguous (4 x 32768 floats)
    float* zreg  = (float*)take((size_t)4 * 32768 * 4);
    float* H0_0  = zreg;
    float* H1_0  = zreg + 32768;
    float* C0    = zreg + 2 * 32768;
    float* C1    = zreg + 3 * 32768;
    float* H0_1  = (float*)take(32768 * 4);
    float* H1_1  = (float*)take(32768 * 4);
    float* H0buf[2] = {H0_0, H0_1};
    float* H1buf[2] = {H1_0, H1_1};

    // ---- repack weights (once per call; deterministic) ----
    const size_t cell0 = (size_t)256 * 64 * 512;   // elems per l0 cell
    const size_t cell1 = (size_t)256 * 96 * 512;
    for (int dir = 0; dir < 2; ++dir) {
        // layer 0: kt 0..31 from w_ih (K=1024), kt 32..63 from w_hh (K=1024)
        {
            int total = 4096 * 1024, grid = (total + 255) / 256;
            pack_w_kernel<<<grid, 256, 0, stream>>>(
                w_ih_l0 + (size_t)dir * 4096 * 1024, Wl0 + dir * cell0, 1024, 64, 0);
            pack_w_kernel<<<grid, 256, 0, stream>>>(
                w_hh_l0 + (size_t)dir * 4096 * 1024, Wl0 + dir * cell0, 1024, 64, 32);
        }
        // layer 1: kt 0..63 from w_ih (K=2048), kt 64..95 from w_hh (K=1024)
        {
            int total = 4096 * 2048, grid = (total + 255) / 256;
            pack_w_kernel<<<grid, 256, 0, stream>>>(
                w_ih_l1 + (size_t)dir * 4096 * 2048, Wl1 + dir * cell1, 2048, 96, 0);
            total = 4096 * 1024; grid = (total + 255) / 256;
            pack_w_kernel<<<grid, 256, 0, stream>>>(
                w_hh_l1 + (size_t)dir * 4096 * 1024, Wl1 + dir * cell1, 1024, 96, 64);
        }
    }
    pack_bias_kernel<<<(8192 + 255) / 256, 256, 0, stream>>>(b_ih_l0, b_hh_l0, bias0, 8192);
    pack_bias_kernel<<<(8192 + 255) / 256, 256, 0, stream>>>(b_ih_l1, b_hh_l1, bias1, 8192);
    zero_kernel<<<(131072 + 255) / 256, 256, 0, stream>>>(zreg, 131072);

    // ---- pipelined scan: one launch per step (513 total) ----
    // Launch t: layer0 computes step t; layer1 computes step t-1.
    //   p = t&1. layer0: read H0[p], write H0[1-p].
    //   layer1 (step t-1): X1 = H0[p] (written at launch t-1 by layer0),
    //                      read H1[1-p], write H1[p].
    const size_t ldsBytes = (size_t)Bd * 3072 * 2 + 4096;  // sized for layer1
    for (int t = 0; t <= Td; ++t) {
        int p = t & 1;
        lstm_pipe_kernel<<<256, 128, ldsBytes, stream>>>(
            inputs, emb, Wl0, Wl1, bias0, bias1,
            H0buf[p], H0buf[1 - p],
            H0buf[p], H1buf[1 - p], H1buf[p],
            C0, C1, out, t);
    }
}
